// DecoderResidual_13322988552309
// MI455X (gfx1250) — matrix operand order repaired and checked by reading, not compiled, not run on hardware
//
#include <hip/hip_runtime.h>
#include <hip/hip_bf16.h>

// Problem constants (from reference)
#define cB 16
#define cS 128
#define cE 512
#define cL 512
#define cH 1024
#define cV 50257
#define cNL 2
#define cG (4 * cH) // 4096 gate width

typedef __attribute__((ext_vector_type(16))) __bf16 v16bf;
typedef __attribute__((ext_vector_type(8)))  float  v8f;

union BF16x16 {
  v16bf v;
  uint4 q[2];
  unsigned short s[16];
};

static __device__ __forceinline__ unsigned short f2b(float f) {
  union { float f; unsigned int u; } x;
  x.f = f;
  unsigned int u = x.u;
  unsigned int r = u + 0x7FFFu + ((u >> 16) & 1u); // RNE
  if ((u & 0x7FFFFFFFu) > 0x7F800000u) r = u | 0x00400000u; // quiet NaN
  return (unsigned short)(r >> 16);
}

static __device__ __forceinline__ float sigm(float x) {
  return 1.0f / (1.0f + __expf(-x));
}

// ---------------------------------------------------------------------------
// Elementwise helpers
// ---------------------------------------------------------------------------

// x[b,s,:] = concat(decoder_input[b,s,:], z[b,:]) ; emit f32 + bf16 copies
__global__ void k_concat(const float* __restrict__ dec, const float* __restrict__ z,
                         float* __restrict__ xf, unsigned short* __restrict__ xb) {
  long long i = (long long)blockIdx.x * blockDim.x + threadIdx.x;
  const long long n = (long long)cB * cS * cH;
  if (i >= n) return;
  int j = (int)(i % cH);
  long long bs = i / cH;
  int s = (int)(bs % cS);
  int b = (int)(bs / cS);
  float v = (j < cE) ? dec[((long long)b * cS + s) * cE + j]
                     : z[(long long)b * cL + (j - cE)];
  xf[i] = v;
  xb[i] = f2b(v);
}

__global__ void k_cvt(const float* __restrict__ src, unsigned short* __restrict__ dst,
                      long long n) {
  long long i = (long long)blockIdx.x * blockDim.x + threadIdx.x;
  if (i < n) dst[i] = f2b(src[i]);
}

__global__ void k_bias_sum(const float* __restrict__ a, const float* __restrict__ b,
                           float* __restrict__ dst, int n) {
  int i = blockIdx.x * blockDim.x + threadIdx.x;
  if (i < n) dst[i] = a[i] + b[i];
}

// residual: of = a + b (f32), ob = bf16(of)
__global__ void k_residual(const float* __restrict__ a, const float* __restrict__ b,
                           float* __restrict__ of, unsigned short* __restrict__ ob,
                           long long n) {
  long long i = (long long)blockIdx.x * blockDim.x + threadIdx.x;
  if (i < n) {
    float v = a[i] + b[i];
    of[i] = v;
    ob[i] = f2b(v);
  }
}

// per-layer state init: h (bf16) and c (f32) from h0/c0 slices
__global__ void k_init_state(const float* __restrict__ h0, const float* __restrict__ c0,
                             unsigned short* __restrict__ hb, float* __restrict__ cs,
                             int n) {
  int i = blockIdx.x * blockDim.x + threadIdx.x;
  if (i < n) {
    hb[i] = f2b(h0[i]);
    cs[i] = c0[i];
  }
}

// copy final states of a layer into d_out: hF from out[:,S-1,:], cF from c state
__global__ void k_state_out(const float* __restrict__ outf, const float* __restrict__ cs,
                            float* __restrict__ hF, float* __restrict__ cF) {
  int i = blockIdx.x * blockDim.x + threadIdx.x;
  if (i >= cB * cH) return;
  int b = i / cH, j = i % cH;
  hF[i] = outf[((long long)b * cS + (cS - 1)) * cH + j];
  cF[i] = cs[i];
}

// ---------------------------------------------------------------------------
// Generic WMMA GEMM: C[M,N] = A[M,K](bf16) @ W[N,K]^T + bias[N]
// WF32: W is f32 (converted in-register), else bf16. Each wave: 16x64 strip.
// ---------------------------------------------------------------------------
template <bool WF32>
__global__ void k_gemm_wmma(const unsigned short* __restrict__ A,
                            const void* __restrict__ Wv,
                            const float* __restrict__ bias,
                            float* __restrict__ C,
                            int M, int N, int K) {
  const int lane = threadIdx.x & 31;
  const int wave = (int)(((long long)blockIdx.x * blockDim.x + threadIdx.x) >> 5);
  const int nstr = (N + 63) >> 6;
  const int mt   = wave / nstr;
  const int ns   = wave % nstr;
  if (mt * 16 >= M) return;
  const int l16 = lane & 15;
  const int hi  = lane >> 4; // 0 or 1
  const int arow = mt * 16 + l16;
  const unsigned short* Wh = (const unsigned short*)Wv;
  const float*          Wf = (const float*)Wv;

  v8f acc[4] = {};
  for (int k0 = 0; k0 < K; k0 += 32) {
    // A tile: lane = M row; 16 contiguous bf16 at k0+16*hi
    BF16x16 a;
    a.q[0] = *(const uint4*)(A + (size_t)arow * K + k0 + hi * 16);
    a.q[1] = *(const uint4*)(A + (size_t)arow * K + k0 + hi * 16 + 8);
#pragma unroll
    for (int j = 0; j < 4; ++j) {
      int ncol = (ns * 4 + j) * 16 + l16;
      int wrow = ncol < N ? ncol : (N - 1); // clamp (guarded at store)
      BF16x16 b;
      if (WF32) {
        const float* src = Wf + (size_t)wrow * K + k0 + hi * 16;
        float4 f0 = *(const float4*)(src + 0);
        float4 f1 = *(const float4*)(src + 4);
        float4 f2 = *(const float4*)(src + 8);
        float4 f3 = *(const float4*)(src + 12);
        b.s[0]  = f2b(f0.x); b.s[1]  = f2b(f0.y); b.s[2]  = f2b(f0.z); b.s[3]  = f2b(f0.w);
        b.s[4]  = f2b(f1.x); b.s[5]  = f2b(f1.y); b.s[6]  = f2b(f1.z); b.s[7]  = f2b(f1.w);
        b.s[8]  = f2b(f2.x); b.s[9]  = f2b(f2.y); b.s[10] = f2b(f2.z); b.s[11] = f2b(f2.w);
        b.s[12] = f2b(f3.x); b.s[13] = f2b(f3.y); b.s[14] = f2b(f3.z); b.s[15] = f2b(f3.w);
      } else {
        const uint4* pw = (const uint4*)(Wh + (size_t)wrow * K + k0 + hi * 16);
        b.q[0] = pw[0];
        b.q[1] = pw[1];
      }
      acc[j] = __builtin_amdgcn_wmma_f32_16x16x32_bf16(
          false, a.v, false, b.v, (short)0, acc[j], false, false);
    }
  }
#pragma unroll
  for (int j = 0; j < 4; ++j) {
    int ncol = (ns * 4 + j) * 16 + l16;
    if (ncol < N) {
      float bv = bias ? bias[ncol] : 0.0f;
#pragma unroll
      for (int r = 0; r < 8; ++r) {
        int m = mt * 16 + r + hi * 8; // C layout: vgpr r -> row r (+8 for hi lanes)
        C[(size_t)m * N + ncol] = acc[j][r] + bv;
      }
    }
  }
}

// ---------------------------------------------------------------------------
// Fused LSTM timestep. Grid: 64 blocks (one per 16-col h tile) x 128 threads
// (4 waves, one per gate i/f/g/o). Each wave: 16x16 gate tile = 32 WMMAs over
// K=H, C seeded from precomputed Xg = x@W_ih^T + b. Gates -> LDS -> cell.
// h double-buffered across launches (hin read-only, hout write-only).
// ---------------------------------------------------------------------------
__global__ void k_lstm_step(const unsigned short* __restrict__ Whh, // [4H, H] bf16
                            const float* __restrict__ Xg,           // [B*S, 4H]
                            const unsigned short* __restrict__ hin, // [B, H] bf16
                            unsigned short* __restrict__ hout,      // [B, H] bf16
                            float* __restrict__ cs,                 // [B, H] f32
                            float* __restrict__ outf,               // [B, S, H]
                            unsigned short* __restrict__ outb,      // [B, S, H]
                            int t) {
  __shared__ float lds[4 * 256];
  const int lane = threadIdx.x & 31;
  const int w    = threadIdx.x >> 5; // gate index 0..3 (i,f,g,o)
  const int nt   = blockIdx.x;       // h-column tile 0..63
  const int l16  = lane & 15;
  const int hi   = lane >> 4;
  const int wrow = w * cH + nt * 16 + l16; // gate-space column == W_hh row

  // Seed accumulator with Xg tile (rows = batch, via C layout)
  v8f acc;
#pragma unroll
  for (int r = 0; r < 8; ++r) {
    int m = r + hi * 8; // batch index
    acc[r] = Xg[((size_t)m * cS + t) * cG + wrow];
  }

  for (int k0 = 0; k0 < cH; k0 += 32) {
    BF16x16 a, b;
    const unsigned short* hrow = hin + (size_t)l16 * cH;
    a.q[0] = *(const uint4*)(hrow + k0 + hi * 16);
    a.q[1] = *(const uint4*)(hrow + k0 + hi * 16 + 8);
    const uint4* pw = (const uint4*)(Whh + (size_t)wrow * cH + k0 + hi * 16);
    b.q[0] = pw[0];
    b.q[1] = pw[1];
    acc = __builtin_amdgcn_wmma_f32_16x16x32_bf16(
        false, a.v, false, b.v, (short)0, acc, false, false);
  }

#pragma unroll
  for (int r = 0; r < 8; ++r) {
    int m = r + hi * 8;
    lds[w * 256 + m * 16 + l16] = acc[r];
  }
  __syncthreads();

  // Cell update: 256 elements (b, n) across 128 threads
  for (int e = threadIdx.x; e < 256; e += 128) {
    int b_  = e >> 4;
    int n   = e & 15;
    int col = nt * 16 + n;
    float gi = lds[0 * 256 + e];
    float gf = lds[1 * 256 + e];
    float gg = lds[2 * 256 + e];
    float go = lds[3 * 256 + e];
    float cold = cs[b_ * cH + col];
    float cn = sigm(gf) * cold + sigm(gi) * tanhf(gg);
    float hn = sigm(go) * tanhf(cn);
    cs[b_ * cH + col] = cn;
    hout[b_ * cH + col] = f2b(hn);
    size_t oi = ((size_t)b_ * cS + t) * cH + col;
    outf[oi] = hn;
    outb[oi] = f2b(hn);
  }
}

// ---------------------------------------------------------------------------
// Host orchestration
// ---------------------------------------------------------------------------
static inline long long cdivll(long long a, long long b) { return (a + b - 1) / b; }

extern "C" void kernel_launch(void* const* d_in, const int* in_sizes, int n_in,
                              void* d_out, int out_size, void* d_ws, size_t ws_size,
                              hipStream_t stream) {
  (void)in_sizes; (void)n_in; (void)out_size;

  const float* dec  = (const float*)d_in[0];
  const float* z    = (const float*)d_in[1];
  // d_in[2] encoder_outputs: unused by forward
  const float* h0   = (const float*)d_in[3];
  const float* c0   = (const float*)d_in[4];
  const float* wih1 = (const float*)d_in[5];
  const float* whh1 = (const float*)d_in[6];
  const float* bih1 = (const float*)d_in[7];
  const float* bhh1 = (const float*)d_in[8];
  const float* wih2 = (const float*)d_in[9];
  const float* whh2 = (const float*)d_in[10];
  const float* bih2 = (const float*)d_in[11];
  const float* bhh2 = (const float*)d_in[12];
  const float* fcw  = (const float*)d_in[13];
  const float* fcb  = (const float*)d_in[14];

  const long long BSH = (long long)cB * cS * cH;      // 2,097,152
  const long long BSG = (long long)cB * cS * cG;      // 8,388,608
  const long long WEL = (long long)cNL * cG * cH;     // 8,388,608 per weight tensor
  const long long FCW = (long long)cV * cH;           // 51,463,168

  // Workspace carve-out
  char* W = (char*)d_ws;
  size_t off = 0;
  auto carve = [&](size_t bytes) -> char* {
    char* p = W + off;
    off += (bytes + 255) & ~(size_t)255;
    return p;
  };
  float*          xf   = (float*)carve(BSH * 4);
  unsigned short* xb   = (unsigned short*)carve(BSH * 2);
  float*          owf  = (float*)carve(BSH * 4);
  unsigned short* owb  = (unsigned short*)carve(BSH * 2);
  float*          oAf  = (float*)carve(BSH * 4);
  unsigned short* oAb  = (unsigned short*)carve(BSH * 2);
  float*          oBf  = (float*)carve(BSH * 4);
  unsigned short* oBb  = (unsigned short*)carve(BSH * 2);
  float*          Xg   = (float*)carve(BSG * 4);
  unsigned short* wih1b = (unsigned short*)carve(WEL * 2);
  unsigned short* whh1b = (unsigned short*)carve(WEL * 2);
  unsigned short* wih2b = (unsigned short*)carve(WEL * 2);
  unsigned short* whh2b = (unsigned short*)carve(WEL * 2);
  float*          bsum1 = (float*)carve((size_t)cNL * cG * 4);
  float*          bsum2 = (float*)carve((size_t)cNL * cG * 4);
  unsigned short* hping = (unsigned short*)carve((size_t)cB * cH * 2);
  unsigned short* hpong = (unsigned short*)carve((size_t)cB * cH * 2);
  float*          cst   = (float*)carve((size_t)cB * cH * 4);

  // Optional: bf16 copy of fc_w (103 MB -> fits 192 MB L2; halves HBM traffic
  // and removes per-tile f32->bf16 VALU converts from the dominant GEMM).
  // Only if the provided workspace is large enough; else fall back to f32 path.
  size_t need_fcwb = off + ((size_t)FCW * 2 + 255);
  unsigned short* fcwb = nullptr;
  if (ws_size >= need_fcwb) {
    fcwb = (unsigned short*)carve(FCW * 2);
  }

  // d_out regions: logits | hF | cF
  float* logits = (float*)d_out;
  float* hF = logits + (long long)cB * cS * cV;
  float* cF = hF + (long long)cNL * cB * cH;

  const int TB = 256;

  // 1) build concatenated input
  k_concat<<<dim3((unsigned)cdivll(BSH, TB)), dim3(TB), 0, stream>>>(dec, z, xf, xb);

  // 2) weights -> bf16 ; combined biases
  k_cvt<<<dim3((unsigned)cdivll(WEL, TB)), dim3(TB), 0, stream>>>(wih1, wih1b, WEL);
  k_cvt<<<dim3((unsigned)cdivll(WEL, TB)), dim3(TB), 0, stream>>>(whh1, whh1b, WEL);
  k_cvt<<<dim3((unsigned)cdivll(WEL, TB)), dim3(TB), 0, stream>>>(wih2, wih2b, WEL);
  k_cvt<<<dim3((unsigned)cdivll(WEL, TB)), dim3(TB), 0, stream>>>(whh2, whh2b, WEL);
  if (fcwb) {
    k_cvt<<<dim3((unsigned)cdivll(FCW, TB)), dim3(TB), 0, stream>>>(fcw, fcwb, FCW);
  }
  k_bias_sum<<<dim3((unsigned)cdivll(cNL * cG, TB)), dim3(TB), 0, stream>>>(bih1, bhh1, bsum1, cNL * cG);
  k_bias_sum<<<dim3((unsigned)cdivll(cNL * cG, TB)), dim3(TB), 0, stream>>>(bih2, bhh2, bsum2, cNL * cG);

  // GEMM launcher (bf16 weights)
  auto gemm_bf16 = [&](const unsigned short* A, const unsigned short* Wgt,
                       const float* bias, float* C, int M, int N, int K) {
    long long nstr = (N + 63) / 64;
    long long waves = (long long)(M / 16) * nstr;
    unsigned blocks = (unsigned)cdivll(waves, 8);
    k_gemm_wmma<false><<<dim3(blocks), dim3(256), 0, stream>>>(A, (const void*)Wgt, bias, C, M, N, K);
  };

  // run one LSTM layer: input_b [B*S,H] bf16, weights slice l
  auto run_layer = [&](const unsigned short* in_b,
                       const unsigned short* wih_b, const unsigned short* whh_b,
                       const float* bsum, int layer,
                       float* outf_, unsigned short* outb_) {
    const unsigned short* wih_l = wih_b + (size_t)layer * cG * cH;
    const unsigned short* whh_l = whh_b + (size_t)layer * cG * cH;
    const float* bsum_l = bsum + (size_t)layer * cG;
    // Xg = in @ W_ih^T + (b_ih + b_hh) for all timesteps at once
    gemm_bf16(in_b, wih_l, bsum_l, Xg, cB * cS, cG, cH);
    // init h, c
    k_init_state<<<dim3((unsigned)cdivll(cB * cH, TB)), dim3(TB), 0, stream>>>(
        h0 + (size_t)layer * cB * cH, c0 + (size_t)layer * cB * cH, hping, cst, cB * cH);
    // sequential recurrence; ping-pong h across launches
    for (int t = 0; t < cS; ++t) {
      const unsigned short* hin = (t & 1) ? hpong : hping;
      unsigned short* hout      = (t & 1) ? hping : hpong;
      k_lstm_step<<<dim3(cH / 16), dim3(128), 0, stream>>>(
          whh_l, Xg, hin, hout, cst, outf_, outb_, t);
    }
  };

  // rnn_1 (2 layers): x -> oA -> oB
  run_layer(xb,  wih1b, whh1b, bsum1, 0, oAf, oAb);
  run_layer(oAb, wih1b, whh1b, bsum1, 1, oBf, oBb);

  // residual: output_words = x + out1
  k_residual<<<dim3((unsigned)cdivll(BSH, TB)), dim3(TB), 0, stream>>>(xf, oBf, owf, owb, BSH);

  // rnn_2 (2 layers): ow -> oA -> oB ; record final states
  run_layer(owb, wih2b, whh2b, bsum2, 0, oAf, oAb);
  k_state_out<<<dim3((unsigned)cdivll(cB * cH, TB)), dim3(TB), 0, stream>>>(
      oAf, cst, hF + 0, cF + 0);
  run_layer(oAb, wih2b, whh2b, bsum2, 1, oBf, oBb);
  k_state_out<<<dim3((unsigned)cdivll(cB * cH, TB)), dim3(TB), 0, stream>>>(
      oBf, cst, hF + (size_t)cB * cH, cF + (size_t)cB * cH);

  // FC head: logits = out2 @ fc_w^T + fc_b
  {
    long long nstr = (cV + 63) / 64;
    long long waves = (long long)(cB * cS / 16) * nstr;
    unsigned blocks = (unsigned)cdivll(waves, 8);
    if (fcwb) {
      k_gemm_wmma<false><<<dim3(blocks), dim3(256), 0, stream>>>(
          oBb, (const void*)fcwb, fcb, logits, cB * cS, cV, cH);
    } else {
      k_gemm_wmma<true><<<dim3(blocks), dim3(256), 0, stream>>>(
          oBb, (const void*)fcw, fcb, logits, cB * cS, cV, cH);
    }
  }
}
